// BDGKT_62354335203406
// MI455X (gfx1250) — matrix operand order repaired and checked by reading, not compiled, not run on hardware
//
#include <hip/hip_runtime.h>

// ---------------- model dims ----------------
#define B_   128
#define U_   50
#define I_   100
#define NU_  6400            // B*U
#define NI_  12800           // B*I
#define D_   64
#define KI_  50
#define KU_  50
#define S_   128
#define L_   2

typedef _Float16 v16h __attribute__((ext_vector_type(16)));
typedef _Float16 v8h  __attribute__((ext_vector_type(8)));
typedef _Float16 v4h  __attribute__((ext_vector_type(4)));
typedef float    v8f  __attribute__((ext_vector_type(8)));

__device__ __forceinline__ float sigmoidf_(float x) { return 1.0f / (1.0f + __expf(-x)); }

// ---------------- WMMA fragment helpers (wave32, 16x16x32 f16) ----------------
// A fragment: 16x32 (MxK) f16. lane: m = l&15, sel = l>>4.
// elements 0..7  = At[m][kOff + sel*8 + 0..7]
// elements 8..15 = At[m][kOff + 16 + sel*8 + 0..7]
__device__ __forceinline__ v16h load_afrag(const _Float16* At, int lane, int kOff) {
    int m = lane & 15, sel = lane >> 4;
    const _Float16* p0 = At + m * 64 + kOff + sel * 8;
    v8h lo = *(const v8h*)p0;
    v8h hi = *(const v8h*)(p0 + 16);
    v16h r;
#pragma unroll
    for (int i = 0; i < 8; ++i) { r[i] = lo[i]; r[8 + i] = hi[i]; }
    return r;
}

// B fragment: 32x16 (KxN) f16 from transposed LDS Wt[n][k] (64x64).
// lane: n = t*16 + (l&15), sel = l>>4; elements 0..7: k = kOff + sel*8 + e, 8..15: k = kOff + 16 + sel*8 + (e-8).
__device__ __forceinline__ v16h load_bfrag(const _Float16* Wt, int lane, int t, int kOff) {
    int n = t * 16 + (lane & 15);
    int kh = (lane >> 4) * 8;
    const _Float16* p = Wt + n * 64 + kOff + kh;
    v8h lo = *(const v8h*)p;
    v8h hi = *(const v8h*)(p + 16);
    v16h r;
#pragma unroll
    for (int i = 0; i < 8; ++i) { r[i] = lo[i]; r[8 + i] = hi[i]; }
    return r;
}

__device__ __forceinline__ v8f wmma_f16(v16h a, v16h b, v8f c) {
    return __builtin_amdgcn_wmma_f32_16x16x32_f16(false, a, false, b, (short)0, c, false, false);
}

// ---- vectorized staging: W (64 rows x 64 cols chunk, row-major, row stride ldw,
//      starting at column nOff) -> Wt[n][k] f16 transposed in LDS.
//      float4 loads along N (coalesced), 4x b16 scatter stores (transpose). ----
__device__ __forceinline__ void stage_wt(const float* __restrict__ W, int ldw, int nOff,
                                         _Float16* Wt, int lane) {
#pragma unroll 4
    for (int i = 0; i < 32; ++i) {          // 1024 float4 quads / 32 lanes
        int q = lane + i * 32;
        int k  = q >> 4;                    // 16 quads per k-row
        int n4 = (q & 15) * 4;
        const float4 w = *(const float4*)(W + (size_t)k * ldw + nOff + n4);
        Wt[(n4 + 0) * 64 + k] = (_Float16)w.x;
        Wt[(n4 + 1) * 64 + k] = (_Float16)w.y;
        Wt[(n4 + 2) * 64 + k] = (_Float16)w.z;
        Wt[(n4 + 3) * 64 + k] = (_Float16)w.w;
    }
}

// ---------------- generic WMMA GEMM: C[M,N] = act( A[M,K] @ W[K,N] + bias (+C) ) ----------------
// grid = (M/16, N/64), block = 32 (one wave). K multiple of 64. W row-major with row stride ldw.
__global__ __launch_bounds__(32) void gemm16_kernel(
    const float* __restrict__ A, int lda,
    const float* __restrict__ W, int ldw,
    const float* __restrict__ bias,
    float* __restrict__ C, int ldc,
    int K, int act, int accum)
{
    __shared__ __align__(16) _Float16 At[16 * 64];
    __shared__ __align__(16) _Float16 Wt[64 * 64];
    int lane = threadIdx.x;
    int bm = blockIdx.x, bn = blockIdx.y;

    v8f acc[4];
#pragma unroll
    for (int t = 0; t < 4; ++t)
#pragma unroll
        for (int v = 0; v < 8; ++v) acc[t][v] = 0.0f;

    for (int kc = 0; kc < K; kc += 64) {
        // A tile: 16x64, vector loads + packed contiguous LDS stores.
#pragma unroll
        for (int i = 0; i < 8; ++i) {       // 256 quads / 32 lanes
            int q = lane + i * 32;
            int m  = q >> 4;
            int k4 = (q & 15) * 4;
            const float4 a = *(const float4*)(A + (size_t)(bm * 16 + m) * lda + kc + k4);
            v4h h; h[0] = (_Float16)a.x; h[1] = (_Float16)a.y;
                   h[2] = (_Float16)a.z; h[3] = (_Float16)a.w;
            *(v4h*)(At + q * 4) = h;        // m*64 + k4 == q*4
        }
        // W chunk transposed into Wt.
        stage_wt(W + (size_t)kc * ldw, ldw, bn * 64, Wt, lane);
        __syncthreads();
#pragma unroll
        for (int ko = 0; ko < 64; ko += 32) {
            v16h a = load_afrag(At, lane, ko);
#pragma unroll
            for (int t = 0; t < 4; ++t)
                acc[t] = wmma_f16(a, load_bfrag(Wt, lane, t, ko), acc[t]);
        }
        __syncthreads();
    }

    int half = lane >> 4, nIdx = lane & 15;
#pragma unroll
    for (int t = 0; t < 4; ++t) {
#pragma unroll
        for (int v = 0; v < 8; ++v) {
            int m = bm * 16 + v + half * 8;
            int n = bn * 64 + t * 16 + nIdx;
            float val = acc[t][v];
            if (bias)  val += bias[n];
            if (accum) val += C[m * ldc + n];
            if (act == 1) val = sigmoidf_(val);
            C[m * ldc + n] = val;
        }
    }
}

// ---------------- simple row gather: dst[r] = src[idx[r]], D=64 ----------------
__global__ __launch_bounds__(64) void gather64_kernel(
    const float* __restrict__ src, const int* __restrict__ idx, float* __restrict__ dst)
{
    int r = blockIdx.x, d = threadIdx.x;
    dst[r * 64 + d] = src[idx[r] * 64 + d];
}

// ---------------- skill + is_cat build ----------------
// is_cat[n][0:64] = item_static[n]; is_cat[n][64:128] = (Q_KC[item_id[n]] @ s_emb)/rowsum
__global__ __launch_bounds__(64) void skill_kernel(
    const float* __restrict__ Q_KC, const float* __restrict__ s_emb,
    const int* __restrict__ item_id, const float* __restrict__ item_static,
    float* __restrict__ is_cat)
{
    int n = blockIdx.x, d = threadIdx.x;
    __shared__ __align__(16) float qrow[S_];
    int it = item_id[n];
    *(float2*)(qrow + d * 2) = *(const float2*)(Q_KC + (size_t)it * S_ + d * 2);
    __syncthreads();
    float acc = 0.0f, ssum = 0.0f;
#pragma unroll 4
    for (int s = 0; s < S_; ++s) {
        float q = qrow[s];
        ssum += q;
        acc  += q * s_emb[s * 64 + d];
    }
    is_cat[n * 128 + d]      = item_static[n * 64 + d];
    is_cat[n * 128 + 64 + d] = acc / ssum;
}

// ---------------- tiny (2 x 64) @ (64 x 64) ----------------
__global__ __launch_bounds__(64) void gemv2_kernel(
    const float* __restrict__ x, const float* __restrict__ W,
    const float* __restrict__ bias, float* __restrict__ out)
{
    int r = blockIdx.x, n = threadIdx.x;
    float acc = bias ? bias[n] : 0.0f;
#pragma unroll 8
    for (int d = 0; d < 64; ++d) acc += x[r * 64 + d] * W[d * 64 + n];
    out[r * 64 + n] = acc;
}

// ---------------- attention over KI=50 entries per item ----------------
__global__ __launch_bounds__(64) void attn_kernel(
    const float* __restrict__ key_msg,  // NU x 64  (ud @ l2w[:D])
    const float* __restrict__ key_abs,  // NI x 64  (item_abs @ l2w[D:2D] + l2_b)
    const float* __restrict__ key_resp, // 2 x 64
    const float* __restrict__ Qm,       // NI x 64
    const float* __restrict__ V_abs,    // NI x 64
    const float* __restrict__ V_resp,   // 2 x 64
    const int*   __restrict__ pby_src,  // NI x 50
    const int*   __restrict__ pby_resp, // NI x 50
    float* __restrict__ item_dyn)       // NI x 64 (out)
{
    int n = blockIdx.x, tid = threadIdx.x;
    __shared__ float Qr[64], kab[64], vab[64];
    __shared__ float krsp[128], vrsp[128];
    __shared__ float e[KI_], alphaL[KI_];
    __shared__ int srcL[KI_], rspL[KI_];

    Qr[tid]  = Qm[n * 64 + tid];
    kab[tid] = key_abs[n * 64 + tid];
    vab[tid] = V_abs[n * 64 + tid];
    krsp[tid] = key_resp[tid];  krsp[64 + tid] = key_resp[64 + tid];
    vrsp[tid] = V_resp[tid];    vrsp[64 + tid] = V_resp[64 + tid];
    if (tid < KI_) {
        srcL[tid] = pby_src[n * KI_ + tid];
        rspL[tid] = pby_resp[n * KI_ + tid];
    }
    __syncthreads();

    if (tid < KI_) {
        int s = srcL[tid], r = rspL[tid];
        float acc = 0.0f;
        const float* km = key_msg + s * 64;
#pragma unroll 8
        for (int d = 0; d < 64; ++d)
            acc += Qr[d] * (km[d] + kab[d] + krsp[r * 64 + d]);
        e[tid] = acc * 0.125f;  // 1/sqrt(64)
    }
    __syncthreads();

    float mx = -1e30f;
    for (int k = 0; k < KI_; ++k) mx = fmaxf(mx, e[k]);
    float sum = 0.0f;
    for (int k = 0; k < KI_; ++k) sum += __expf(e[k] - mx);
    if (tid < KI_) alphaL[tid] = __expf(e[tid] - mx) / sum;
    __syncthreads();

    float out = 0.0f;
#pragma unroll 2
    for (int k = 0; k < KI_; ++k)
        out += alphaL[k] * (vab[tid] + vrsp[rspL[k] * 64 + tid]);
    item_dyn[n * 64 + tid] = out;
}

// ---------------- recurrent scan: 16 independent rows per block, 50 steps, WMMA per step ----------------
__global__ __launch_bounds__(32) void scan_kernel(
    const float* __restrict__ q1sum,   // NI x 64 : idn@q1w[:D] + item_abs@q1w[D:2D] + q1_b
    const int*   __restrict__ by_src,  // NU x 50
    const int*   __restrict__ by_resp, // NU x 50
    const float* __restrict__ lo_resp, // 2 x 64 (incl Lo_b)
    const float* __restrict__ fo_resp, // 2 x 64 (incl fo_b)
    const float* __restrict__ k_init,  // 64
    const float* __restrict__ q1w_k,   // 64x64 (q1_w rows 2D:)
    const float* __restrict__ lqw,     // 64x64
    const float* __restrict__ lqb,     // 64
    const float* __restrict__ low_q,   // 64x64 (Lo_w rows :D)
    const float* __restrict__ fow_k,   // 64x64 (fo_w rows D:)
    float* __restrict__ user_dyn)      // NU x 64 (out)
{
    __shared__ __align__(16) _Float16 Wq1[64 * 64], Wlq[64 * 64], Wlo[64 * 64], Wfo[64 * 64];
    __shared__ __align__(16) _Float16 kkkA[16 * 64], q1A[16 * 64];
    __shared__ int   srcL[16 * KU_], respL[16 * KU_];
    __shared__ float loL[128], foL[128], lqbL[64];

    int lane = threadIdx.x;
    int row0 = blockIdx.x * 16;

    stage_wt(q1w_k, 64, 0, Wq1, lane);
    stage_wt(lqw,   64, 0, Wlq, lane);
    stage_wt(low_q, 64, 0, Wlo, lane);
    stage_wt(fow_k, 64, 0, Wfo, lane);
    for (int idx = lane; idx < 16 * KU_; idx += 32) {
        int m = idx / KU_, k = idx % KU_;
        srcL[idx]  = by_src[(row0 + m) * KU_ + k];
        respL[idx] = by_resp[(row0 + m) * KU_ + k];
    }
    for (int idx = lane; idx < 128; idx += 32) { loL[idx] = lo_resp[idx]; foL[idx] = fo_resp[idx]; }
    for (int idx = lane; idx < 64; idx += 32) lqbL[idx] = lqb[idx];

    int nIdx = lane & 15, half = lane >> 4;
    float kkk[4][8];
#pragma unroll
    for (int t = 0; t < 4; ++t)
#pragma unroll
        for (int v = 0; v < 8; ++v) kkk[t][v] = k_init[t * 16 + nIdx];
#pragma unroll
    for (int t = 0; t < 4; ++t)
#pragma unroll
        for (int v = 0; v < 8; ++v)
            kkkA[(v + half * 8) * 64 + t * 16 + nIdx] = (_Float16)kkk[t][v];
    __syncthreads();

    for (int step = 0; step < KU_; ++step) {
        // q1_acc = kkk @ q1w_k ; fo_acc = kkk @ fow_k
        v8f accQ[4], accF[4];
#pragma unroll
        for (int t = 0; t < 4; ++t)
#pragma unroll
            for (int v = 0; v < 8; ++v) { accQ[t][v] = 0.0f; accF[t][v] = 0.0f; }
#pragma unroll
        for (int ko = 0; ko < 64; ko += 32) {
            v16h a = load_afrag(kkkA, lane, ko);
#pragma unroll
            for (int t = 0; t < 4; ++t) {
                accQ[t] = wmma_f16(a, load_bfrag(Wq1, lane, t, ko), accQ[t]);
                accF[t] = wmma_f16(a, load_bfrag(Wfo, lane, t, ko), accF[t]);
            }
        }
        // epilogue A: q1 = acc + q1sum[by_src[m,step]][d]; foin = sigmoid(acc + fo_pre)
        float foin[4][8];
#pragma unroll
        for (int t = 0; t < 4; ++t) {
#pragma unroll
            for (int v = 0; v < 8; ++v) {
                int m = v + half * 8;
                int d = t * 16 + nIdx;
                int src = srcL[m * KU_ + step];
                int rsp = respL[m * KU_ + step];
                float q1 = accQ[t][v] + q1sum[src * 64 + d];
                q1A[m * 64 + d] = (_Float16)q1;
                foin[t][v] = sigmoidf_(accF[t][v] + foL[rsp * 64 + d]);
            }
        }
        __syncthreads();
        // lq_acc = q1 @ lqw ; lo_acc = q1 @ low_q
        v8f accL[4], accS[4];
#pragma unroll
        for (int t = 0; t < 4; ++t)
#pragma unroll
            for (int v = 0; v < 8; ++v) { accL[t][v] = 0.0f; accS[t][v] = 0.0f; }
#pragma unroll
        for (int ko = 0; ko < 64; ko += 32) {
            v16h a = load_afrag(q1A, lane, ko);
#pragma unroll
            for (int t = 0; t < 4; ++t) {
                accL[t] = wmma_f16(a, load_bfrag(Wlq, lane, t, ko), accL[t]);
                accS[t] = wmma_f16(a, load_bfrag(Wlo, lane, t, ko), accS[t]);
            }
        }
#pragma unroll
        for (int t = 0; t < 4; ++t) {
#pragma unroll
            for (int v = 0; v < 8; ++v) {
                int m = v + half * 8;
                int d = t * 16 + nIdx;
                int rsp = respL[m * KU_ + step];
                float xt = tanhf(accL[t][v] + lqbL[d]);
                float xx = sigmoidf_(accS[t][v] + loL[rsp * 64 + d]) * xt;
                float f = foin[t][v];
                kkk[t][v] = f * kkk[t][v] + (1.0f - f) * xx;
            }
        }
        __syncthreads();
#pragma unroll
        for (int t = 0; t < 4; ++t)
#pragma unroll
            for (int v = 0; v < 8; ++v)
                kkkA[(v + half * 8) * 64 + t * 16 + nIdx] = (_Float16)kkk[t][v];
        __syncthreads();
    }

#pragma unroll
    for (int t = 0; t < 4; ++t)
#pragma unroll
        for (int v = 0; v < 8; ++v)
            user_dyn[(row0 + v + half * 8) * 64 + t * 16 + nIdx] = kkk[t][v];
}

// ---------------- per-layer output collection ----------------
__global__ __launch_bounds__(64) void collect_kernel(
    const float* __restrict__ user_dyn, const float* __restrict__ item_dyn,
    const float* __restrict__ t_item_static,
    const int* __restrict__ user_index, const int* __restrict__ item_index,
    const int* __restrict__ exist,
    float* __restrict__ user_layer, float* __restrict__ item_layer, int layer)
{
    int b = blockIdx.x, d = threadIdx.x;
    int u = b * U_ + user_index[b];
    int i = b * I_ + item_index[b];
    user_layer[b * 192 + layer * 64 + d] = user_dyn[u * 64 + d];
    float iv = item_dyn[i * 64 + d];
    if (layer > 0 && exist[b] == 0) iv = t_item_static[b * 64 + d];
    item_layer[b * 192 + layer * 64 + d] = iv;
}

// ---------------- build concat([user_01, 1-RD]) ----------------
__global__ __launch_bounds__(128) void catdf_kernel(
    const float* __restrict__ u01, const float* __restrict__ RD, float* __restrict__ catb)
{
    int b = blockIdx.x, s = threadIdx.x;
    catb[b * 256 + s]       = u01[b * 128 + s];
    catb[b * 256 + 128 + s] = 1.0f - RD[b * 128 + s];
}

// ---------------- final masked reduction ----------------
__global__ __launch_bounds__(128) void final_kernel(
    const float* __restrict__ DF, const float* __restrict__ diff,
    const float* __restrict__ Q_KC, const int* __restrict__ titem_id,
    float* __restrict__ out)
{
    int b = blockIdx.x, s = threadIdx.x;
    __shared__ float pa[128], sa[128];
    float t = Q_KC[titem_id[b] * S_ + s];
    // sigmoid(-5*(DF-diff)) = 1/(1+exp(5*(DF-diff)))
    float P = 1.0f / (1.0f + __expf(5.0f * (DF[b * 128 + s] - diff[b * 128 + s])));
    pa[s] = P * t;
    sa[s] = t;
    __syncthreads();
    for (int off = 64; off > 0; off >>= 1) {
        if (s < off) { pa[s] += pa[s + off]; sa[s] += sa[s + off]; }
        __syncthreads();
    }
    if (s == 0) out[b] = pa[0] / sa[0];
}

// ---------------- host orchestration ----------------
extern "C" void kernel_launch(void* const* d_in, const int* in_sizes, int n_in,
                              void* d_out, int out_size, void* d_ws, size_t ws_size,
                              hipStream_t stream) {
    (void)in_sizes; (void)n_in; (void)out_size; (void)ws_size;

    const int* user_id    = (const int*)d_in[0];
    const int* item_id    = (const int*)d_in[1];
    const int* pby_src    = (const int*)d_in[2];
    const int* pby_resp   = (const int*)d_in[3];
    const int* by_src     = (const int*)d_in[4];
    const int* by_resp    = (const int*)d_in[5];
    // d_in[6] = tuser_id (unused by forward)
    const int* titem_id   = (const int*)d_in[7];
    const int* user_index = (const int*)d_in[8];
    const int* item_index = (const int*)d_in[9];
    const int* exist      = (const int*)d_in[10];
    const float* u_emb  = (const float*)d_in[11];
    const float* q_emb  = (const float*)d_in[12];
    const float* s_emb  = (const float*)d_in[13];
    const float* r_emb  = (const float*)d_in[14];
    const float* Q_KC   = (const float*)d_in[15];
    const float* Wu     = (const float*)d_in[16];
    const float* Wi     = (const float*)d_in[17];
    const float* k_init = (const float*)d_in[18];
    const float* q1_w   = (const float*)d_in[19];
    const float* q1_b   = (const float*)d_in[20];
    const float* Lo_w   = (const float*)d_in[21];
    const float* Lo_b   = (const float*)d_in[22];
    const float* Lq_w   = (const float*)d_in[23];
    const float* Lq_b   = (const float*)d_in[24];
    const float* fo_w   = (const float*)d_in[25];
    const float* fo_b   = (const float*)d_in[26];
    const float* l1_w   = (const float*)d_in[27];
    const float* l1_b   = (const float*)d_in[28];
    const float* l2_w   = (const float*)d_in[29];
    const float* l2_b   = (const float*)d_in[30];
    const float* l3_w   = (const float*)d_in[31];
    const float* l3_b   = (const float*)d_in[32];
    const float* l4_w   = (const float*)d_in[33];
    const float* l4_b   = (const float*)d_in[34];
    const float* uu_w   = (const float*)d_in[35];
    const float* uu_b   = (const float*)d_in[36];
    const float* ui_w   = (const float*)d_in[37];
    const float* ui_b   = (const float*)d_in[38];
    const float* tdu_w  = (const float*)d_in[39];
    const float* tdu_b  = (const float*)d_in[40];
    const float* tdi_w  = (const float*)d_in[41];
    const float* tdi_b  = (const float*)d_in[42];
    const float* tds_w  = (const float*)d_in[43];
    const float* tds_b  = (const float*)d_in[44];
    const float* tad_w  = (const float*)d_in[45];
    const float* tad_b  = (const float*)d_in[46];

    // workspace carve-up (floats)
    float* ws = (float*)d_ws;
    size_t off = 0;
    float* user_dyn  = ws + off; off += (size_t)NU_ * 64;
    float* item_dyn  = ws + off; off += (size_t)NI_ * 64;
    float* t_item_st = ws + off; off += (size_t)B_ * 64;
    float* is_cat    = ws + off; off += (size_t)NI_ * 128;
    float* ud        = ws + off; off += (size_t)NU_ * 64;
    float* idn       = ws + off; off += (size_t)NI_ * 64;
    float* item_abs  = ws + off; off += (size_t)NI_ * 64;
    float* key_msg   = ws + off; off += (size_t)NU_ * 64;
    float* key_abs   = ws + off; off += (size_t)NI_ * 64;
    float* key_resp  = ws + off; off += 2 * 64;
    float* Qm        = ws + off; off += (size_t)NI_ * 64;
    float* V_abs     = ws + off; off += (size_t)NI_ * 64;
    float* V_resp    = ws + off; off += 2 * 64;
    float* q1sum     = ws + off; off += (size_t)NI_ * 64;
    float* lo_resp   = ws + off; off += 2 * 64;
    float* fo_resp   = ws + off; off += 2 * 64;
    float* user_lyr  = ws + off; off += (size_t)B_ * 192;
    float* item_lyr  = ws + off; off += (size_t)B_ * 192;
    float* t_user    = ws + off; off += (size_t)B_ * 64;
    float* t_item    = ws + off; off += (size_t)B_ * 64;
    float* RD        = ws + off; off += (size_t)B_ * 128;
    float* U01       = ws + off; off += (size_t)B_ * 128;
    float* catb      = ws + off; off += (size_t)B_ * 256;
    float* DF        = ws + off; off += (size_t)B_ * 128;
    float* diffb     = ws + off; off += (size_t)B_ * 128;

    // ---- static gathers + skill ----
    gather64_kernel<<<NU_, 64, 0, stream>>>(u_emb, user_id, user_dyn);
    gather64_kernel<<<NI_, 64, 0, stream>>>(q_emb, item_id, item_dyn);
    gather64_kernel<<<B_,  64, 0, stream>>>(q_emb, titem_id, t_item_st);
    skill_kernel<<<NI_, 64, 0, stream>>>(Q_KC, s_emb, item_id, item_dyn, is_cat);
    collect_kernel<<<B_, 64, 0, stream>>>(user_dyn, item_dyn, t_item_st,
                                          user_index, item_index, exist,
                                          user_lyr, item_lyr, 0);

    const int DD = 64 * 64;  // one D x D matrix
    for (int l = 0; l < L_; ++l) {
        const float* Wu_l  = Wu + l * DD;
        const float* Wi_l  = Wi + l * DD;
        const float* l1w_l = l1_w + l * 2 * DD;  const float* l1b_l = l1_b + l * 64;
        const float* l2w_l = l2_w + l * 3 * DD;  const float* l2b_l = l2_b + l * 64;
        const float* l3w_l = l3_w + l * DD;      const float* l3b_l = l3_b + l * 64;
        const float* l4w_l = l4_w + l * 2 * DD;  const float* l4b_l = l4_b + l * 64;
        const float* q1w_l = q1_w + l * 3 * DD;  const float* q1b_l = q1_b + l * 64;
        const float* low_l = Lo_w + l * 2 * DD;  const float* lob_l = Lo_b + l * 64;
        const float* lqw_l = Lq_w + l * DD;      const float* lqb_l = Lq_b + l * 64;
        const float* fow_l = fo_w + l * 2 * DD;  const float* fob_l = fo_b + l * 64;
        const float* ki_l  = k_init + l * 64;

        gemm16_kernel<<<dim3(NU_ / 16, 1), 32, 0, stream>>>(user_dyn, 64, Wu_l, 64, nullptr, ud, 64, 64, 0, 0);
        gemm16_kernel<<<dim3(NI_ / 16, 1), 32, 0, stream>>>(item_dyn, 64, Wi_l, 64, nullptr, idn, 64, 64, 0, 0);
        gemm16_kernel<<<dim3(NI_ / 16, 1), 32, 0, stream>>>(is_cat, 128, l1w_l, 64, l1b_l, item_abs, 64, 128, 0, 0);
        gemm16_kernel<<<dim3(NU_ / 16, 1), 32, 0, stream>>>(ud, 64, l2w_l, 64, nullptr, key_msg, 64, 64, 0, 0);
        gemm16_kernel<<<dim3(NI_ / 16, 1), 32, 0, stream>>>(item_abs, 64, l2w_l + DD, 64, l2b_l, key_abs, 64, 64, 0, 0);
        gemv2_kernel<<<2, 64, 0, stream>>>(r_emb, l2w_l + 2 * DD, nullptr, key_resp);
        gemm16_kernel<<<dim3(NI_ / 16, 1), 32, 0, stream>>>(item_abs, 64, l3w_l, 64, l3b_l, Qm, 64, 64, 0, 0);
        gemm16_kernel<<<dim3(NI_ / 16, 1), 32, 0, stream>>>(item_abs, 64, l4w_l, 64, l4b_l, V_abs, 64, 64, 0, 0);
        gemv2_kernel<<<2, 64, 0, stream>>>(r_emb, l4w_l + DD, nullptr, V_resp);

        // q1sum = idn @ q1w[:D] + q1_b, then += item_abs @ q1w[D:2D]
        gemm16_kernel<<<dim3(NI_ / 16, 1), 32, 0, stream>>>(idn, 64, q1w_l, 64, q1b_l, q1sum, 64, 64, 0, 0);
        gemm16_kernel<<<dim3(NI_ / 16, 1), 32, 0, stream>>>(item_abs, 64, q1w_l + DD, 64, nullptr, q1sum, 64, 64, 0, 1);

        attn_kernel<<<NI_, 64, 0, stream>>>(key_msg, key_abs, key_resp, Qm, V_abs, V_resp,
                                            pby_src, pby_resp, item_dyn);

        gemv2_kernel<<<2, 64, 0, stream>>>(r_emb, low_l + DD, lob_l, lo_resp);
        gemv2_kernel<<<2, 64, 0, stream>>>(r_emb, fow_l, fob_l, fo_resp);

        scan_kernel<<<NU_ / 16, 32, 0, stream>>>(q1sum, by_src, by_resp, lo_resp, fo_resp,
                                                 ki_l, q1w_l + 2 * DD, lqw_l, lqb_l,
                                                 low_l, fow_l + DD, user_dyn);

        collect_kernel<<<B_, 64, 0, stream>>>(user_dyn, item_dyn, t_item_st,
                                              user_index, item_index, exist,
                                              user_lyr, item_lyr, l + 1);
    }

    // ---- heads ----
    gemm16_kernel<<<dim3(B_ / 16, 1), 32, 0, stream>>>(user_lyr, 192, uu_w, 64, uu_b, t_user, 64, 192, 0, 0);
    gemm16_kernel<<<dim3(B_ / 16, 1), 32, 0, stream>>>(item_lyr, 192, ui_w, 64, ui_b, t_item, 64, 192, 0, 0);
    gemm16_kernel<<<dim3(B_ / 16, 2), 32, 0, stream>>>(t_item, 64, tdi_w, 128, tdi_b, RD, 128, 64, 1, 0);
    gemm16_kernel<<<dim3(B_ / 16, 2), 32, 0, stream>>>(t_user, 64, tdu_w, 128, tdu_b, U01, 128, 64, 1, 0);
    catdf_kernel<<<B_, 128, 0, stream>>>(U01, RD, catb);
    gemm16_kernel<<<dim3(B_ / 16, 2), 32, 0, stream>>>(catb, 256, tds_w, 128, tds_b, DF, 128, 256, 1, 0);
    gemm16_kernel<<<dim3(B_ / 16, 2), 32, 0, stream>>>(t_item_st, 64, tad_w, 128, tad_b, diffb, 128, 64, 1, 0);
    final_kernel<<<B_, 128, 0, stream>>>(DF, diffb, Q_KC, titem_id, (float*)d_out);
}
